// ScaleDotProductAttention_48266842472927
// MI455X (gfx1250) — matrix operand order repaired and checked by reading, not compiled, not run on hardware
//
#include <hip/hip_runtime.h>

// ---------------------------------------------------------------------------
// Scaled dot-product attention for gfx1250 (MI455X), wave32 + WMMA.
//   out[b,h,q,d]  = softmax( mask? (QK^T)/8 : -1e-12 , axis=k ) @ V
//   d_out layout: [ out (B*H*S*D f32) | p_attention (B*H*S*S f32) ]
//
// One wave handles one 16-row Q tile; the 8 waves of a block share one (b,h).
// Pass 1: S^T = K·Q^T via WMMA (K = A operand, Q = loop-invariant B operand)
//   -> each softmax row lives inside a single lane pair (online max/sum is
//      in-lane + one shfl_xor(16)); raw masked scores stashed to the
//      p_attention buffer with contiguous v8f stores.
// Pass 2: reload raw scores (L2-resident), normalize, write final
//   p_attention, and feed P straight back as the WMMA A operand (the S^T
//   C-layout matches the A layout exactly across an even/odd tile pair).
//   V chunks are cooperatively staged per-block into LDS as f16, TRANSPOSED
//   (vt[d][key], stride 40 halves -> 16B-aligned, bank-spread rows), so the
//   B operand is two ds_load_b128 per d-column instead of 64 strided dwords.
// ---------------------------------------------------------------------------

typedef __attribute__((ext_vector_type(16))) _Float16 v16h;
typedef __attribute__((ext_vector_type(8)))  _Float16 v8h;
typedef __attribute__((ext_vector_type(8)))  float    v8f;
typedef __attribute__((ext_vector_type(4)))  float    v4f;

#define WMMA_F16(a, b, c) \
  __builtin_amdgcn_wmma_f32_16x16x32_f16(false, (a), false, (b), (short)0, (c), false, false)

__global__ __launch_bounds__(256) void
ScaleDotProductAttention_48266842472927_kernel(const float* __restrict__ Q,
                                               const float* __restrict__ K,
                                               const float* __restrict__ V,
                                               const int*   __restrict__ M,
                                               float* __restrict__ Out,
                                               float* __restrict__ P)
{
    constexpr int   S      = 2048;
    constexpr int   D      = 64;
    constexpr float SCALE  = 0.125f;        // 1/sqrt(64)
    constexpr float NEG_E  = -1e-12f;       // faithful masked fill (not -inf)
    constexpr int   VSTRIDE = 40;           // halves per LDS row (80B, 16B-aligned)

    // Double-buffered transposed f16 V chunk: vt[buf][d][key], d=0..63, key=0..31
    __shared__ __align__(16) _Float16 vlds[2][64 * VSTRIDE];

    const int tid  = threadIdx.x;
    const int lane = tid & 31;
    const int wave = tid >> 5;
    const int tile = blockIdx.x * 8 + wave;   // 4096 q-tiles total
    const int bh   = tile >> 7;               // 128 q-tiles per (b,h); same for whole block
    const int qt   = tile & 127;
    const int q0   = qt * 16;
    const int b    = bh >> 4;                 // H = 16

    const int nl = lane & 15;                 // q column within tile
    const int hi = lane >> 4;                 // half-wave selector

    const float* Qb = Q   + (size_t)bh * S * D;
    const float* Kb = K   + (size_t)bh * S * D;
    const float* Vb = V   + (size_t)bh * S * D;
    const int*   Mb = M   + (size_t)b  * S * S;
    float*       Pb = P   + (size_t)bh * S * S;
    float*       Ob = Out + (size_t)bh * S * D;

    const int    q     = q0 + nl;             // this lane's softmax row
    const size_t qrowP = (size_t)q * S;

    // ---- Q as WMMA B operand (loaded & converted once; invariant over keys)
    // B layout: element t -> Kdim = t + 16*hi, N = lane%16.
    v16h bq[2];
    {
        const float* qrow = Qb + (size_t)q * D;
#pragma unroll
        for (int c = 0; c < 2; ++c) {
            const float* src = qrow + c * 32 + 16 * hi; // 16 contiguous f32
#pragma unroll
            for (int t = 0; t < 16; ++t) bq[c][t] = (_Float16)src[t];
        }
    }

    // =======================================================================
    // Pass 1: S^T = K·Q^T, mask+scale, store raw scores, online max/sum.
    // =======================================================================
    float m_run = -3.402823466e+38f;
    float l_run = 0.0f;

    for (int kt = 0; kt < 128; ++kt) {
        // K tile as A operand. A layout: lane row = key kt*16 + lane%16,
        // element t -> d = c*32 + t + 16*hi  (one 64B run).
        const float* krow = Kb + (size_t)(kt * 16 + nl) * D;
        if (kt < 127)
            __builtin_prefetch(Kb + (size_t)((kt + 1) * 16 + nl) * D, 0, 1);

        v16h ak[2];
#pragma unroll
        for (int c = 0; c < 2; ++c) {
            const float* s0 = krow + c * 32 + 16 * hi;
#pragma unroll
            for (int t = 0; t < 8; ++t) ak[c][t]     = (_Float16)s0[t];
#pragma unroll
            for (int t = 0; t < 8; ++t) ak[c][8 + t] = (_Float16)s0[8 + t];
        }

        v8f c8 = {};
        c8 = WMMA_F16(ak[0], bq[0], c8);   // d = 0..31
        c8 = WMMA_F16(ak[1], bq[1], c8);   // d = 32..63

        // C layout (S^T): vgpr r -> key kt*16 + 8*hi + r, this lane's q.
        const int  key0 = kt * 16 + 8 * hi;
        const int* mrow = Mb + qrowP + key0;   // 8 contiguous ints per lane
        v8f s;
#pragma unroll
        for (int r = 0; r < 8; ++r)
            s[r] = (mrow[r] != 0) ? c8[r] * SCALE : NEG_E;

        // stash raw masked scores (32B contiguous, 32B aligned)
        *(v8f*)(Pb + qrowP + key0) = s;

        // online softmax over this lane's 8 keys
        float tmax = s[0];
#pragma unroll
        for (int r = 1; r < 8; ++r) tmax = fmaxf(tmax, s[r]);
        const float nm = fmaxf(m_run, tmax);
        float tsum = 0.0f;
#pragma unroll
        for (int r = 0; r < 8; ++r) tsum += __expf(s[r] - nm);
        l_run = l_run * __expf(m_run - nm) + tsum;
        m_run = nm;
    }

    // combine the two half-rows of each q (lane <-> lane^16)
    const float mo    = __shfl_xor(m_run, 16, 32);
    const float lo    = __shfl_xor(l_run, 16, 32);
    const float mf    = fmaxf(m_run, mo);
    const float lf    = l_run * __expf(m_run - mf) + lo * __expf(mo - mf);
    const float inv_l = 1.0f / lf;

    // =======================================================================
    // Pass 2: normalize -> write p_attention, and O = P·V via WMMA.
    // V chunks staged cooperatively per block (all 8 waves share bh).
    // =======================================================================
    v8f o[4] = { v8f{}, v8f{}, v8f{}, v8f{} };

    // staging role for this thread: key row + 8-wide d slice
    const int skey = tid >> 3;          // 0..31
    const int sd0  = (tid & 7) * 8;     // 0,8,...,56

    for (int kt2 = 0; kt2 < 64; ++kt2) {
        const int buf = kt2 & 1;

        // ---- cooperative stage: V[kt2*32 + skey][sd0..sd0+7] -> f16,
        //      transposed into vlds[buf][d][key]  (fully coalesced b128 loads)
        {
            const float* vsrc = Vb + (size_t)(kt2 * 32 + skey) * D + sd0;
            v4f x0 = *(const v4f*)(vsrc);
            v4f x1 = *(const v4f*)(vsrc + 4);
            _Float16* dst = &vlds[buf][0];
#pragma unroll
            for (int i = 0; i < 4; ++i)
                dst[(sd0 + i) * VSTRIDE + skey] = (_Float16)x0[i];
#pragma unroll
            for (int i = 0; i < 4; ++i)
                dst[(sd0 + 4 + i) * VSTRIDE + skey] = (_Float16)x1[i];
        }
        __syncthreads();   // uniform trip count -> safe; also fences prior buf reads

        // ---- P tile pair: normalize + write final p_attention, build A operand
        v16h ap;
#pragma unroll
        for (int sub = 0; sub < 2; ++sub) {
            const size_t off  = qrowP + (size_t)(kt2 * 32 + sub * 16 + 8 * hi);
            const v8f    sraw = *(const v8f*)(Pb + off);
            v8f          pr;
#pragma unroll
            for (int r = 0; r < 8; ++r) pr[r] = __expf(sraw[r] - mf) * inv_l;
            *(v8f*)(Pb + off) = pr;                    // final p_attention
#pragma unroll
            for (int r = 0; r < 8; ++r) ap[sub * 8 + r] = (_Float16)pr[r];
        }

        // ---- V as B operand from LDS: element t -> key = (t<8 ? t : t+8) + 8*hi, col d.
        //      Row d is contiguous: two 16B ds loads per d-column.
#pragma unroll
        for (int dc = 0; dc < 4; ++dc) {
            const _Float16* row = &vlds[buf][(dc * 16 + nl) * VSTRIDE + 8 * hi];
            union { v16h v; v8h h[2]; } u;
            u.h[0] = *(const v8h*)(row);
            u.h[1] = *(const v8h*)(row + 16);
            o[dc] = WMMA_F16(ap, u.v, o[dc]);
        }
    }

    // O is in C layout: vgpr r -> q = q0 + r + 8*hi, col = dc*16 + lane%16.
#pragma unroll
    for (int dc = 0; dc < 4; ++dc)
#pragma unroll
        for (int r = 0; r < 8; ++r)
            Ob[(size_t)(q0 + r + 8 * hi) * D + dc * 16 + nl] = o[dc][r];
}

extern "C" void kernel_launch(void* const* d_in, const int* in_sizes, int n_in,
                              void* d_out, int out_size, void* d_ws, size_t ws_size,
                              hipStream_t stream) {
    (void)in_sizes; (void)n_in; (void)out_size; (void)d_ws; (void)ws_size;

    constexpr int B = 2, H = 16, S = 2048, D = 64;
    const float* Q = (const float*)d_in[0];
    const float* K = (const float*)d_in[1];
    const float* V = (const float*)d_in[2];
    const int*   M = (const int*)  d_in[3];

    float* Out = (float*)d_out;                         // [B,H,S,D]
    float* P   = (float*)d_out + (size_t)B * H * S * D; // [B,H,S,S]

    const int tiles  = B * H * (S / 16);  // 4096 waves
    const int blocks = tiles / 8;         // 256 threads = 8 waves per block
    ScaleDotProductAttention_48266842472927_kernel<<<blocks, 256, 0, stream>>>(
        Q, K, V, M, Out, P);
}
